// TreeGradient_5927054868985
// MI455X (gfx1250) — compile-verified, matrix operand order checked by reading, not run on hardware
//
#include <hip/hip_runtime.h>
#include <hip/hip_bf16.h>
#include <stdint.h>

// Problem constants
#define Nn   1024
#define Mm   1152
#define Ll   32
#define K2   2304          // 2*Mm, fused K dimension [r | h]
#define KC   129           // conv2 taps
#define OUTW 1024          // Mm - KC + 1

// GEMM tiling
#define BM   128
#define BN   128
#define BK   32
#define PITCH (BK + 8)     // bf16 elements per LDS row (80B pitch, 16B-aligned)

typedef __attribute__((ext_vector_type(16))) __bf16 v16bf;
typedef __attribute__((ext_vector_type(8)))  float  v8f;
typedef int v2i_ __attribute__((ext_vector_type(2)));

#if __has_builtin(__builtin_amdgcn_global_load_async_to_lds_b64)
#define ASYNC_LDS 1
#else
#define ASYNC_LDS 0
#endif

// ---------- helpers ----------
__device__ __forceinline__ uint32_t pack2bf(float a, float b) {
#if __has_builtin(__builtin_amdgcn_cvt_pk_bf16_f32)
    auto v = __builtin_amdgcn_cvt_pk_bf16_f32(a, b);   // v_cvt_pk_bf16_f32
    return __builtin_bit_cast(uint32_t, v);
#else
    uint32_t ua = __builtin_bit_cast(uint32_t, a);
    uint32_t ub = __builtin_bit_cast(uint32_t, b);
    ua = ua + 0x7FFFu + ((ua >> 16) & 1u);
    ub = ub + 0x7FFFu + ((ub >> 16) & 1u);
    return (ua >> 16) | (ub & 0xFFFF0000u);            // lo = bf16(a), hi = bf16(b)
#endif
}
__device__ __forceinline__ __bf16 f2bf(float f) {
    uint32_t u = __builtin_bit_cast(uint32_t, f);
    u = u + 0x7FFFu + ((u >> 16) & 1u);
    unsigned short h = (unsigned short)(u >> 16);
    return __builtin_bit_cast(__bf16, h);
}
// Load one 16-bit-element WMMA fragment: 16B at p, 16B at p+16 elements.
__device__ __forceinline__ v16bf ld_frag(const __bf16* p) {
    union { uint4 u[2]; v16bf v; } r;
    r.u[0] = *(const uint4*)(p);
    r.u[1] = *(const uint4*)(p + 16);
    return r.v;
}
#if ASYNC_LDS
__device__ __forceinline__ void async_cp_b64(const void* g, void* l) {
    // Param types are V2i addrspace(1)* / V2i addrspace(3)*.
    // Go through integers so the casts are always legal; generic->LDS is a
    // low-32-bit truncation per the gfx1250 aperture rules.
    __builtin_amdgcn_global_load_async_to_lds_b64(
        (v2i_ __attribute__((address_space(1)))*)(uintptr_t)g,
        (v2i_ __attribute__((address_space(3)))*)(uint32_t)(uintptr_t)l,
        0, 0);
}
__device__ __forceinline__ void wait_async0() {
#if __has_builtin(__builtin_amdgcn_s_wait_asynccnt)
    __builtin_amdgcn_s_wait_asynccnt(0);
#else
    asm volatile("s_wait_asynccnt 0x0" ::: "memory");
#endif
}
#endif

// ---------- 1) weight prep: Wb[j][k] (bf16), k<Mm -> W0[j][k], k>=Mm -> W1[j][k-Mm] ----------
__global__ __launch_bounds__(256) void prep_w(const float* __restrict__ convW,
                                              __bf16* __restrict__ Wb) {
    int idx = blockIdx.x * 256 + threadIdx.x;     // over Mm*Mm
    if (idx >= Mm * Mm) return;
    int j = idx / Mm, k = idx % Mm;
    float w0 = convW[(size_t)idx * 2 + 0];
    float w1 = convW[(size_t)idx * 2 + 1];
    Wb[(size_t)j * K2 + k]      = f2bf(w0);
    Wb[(size_t)j * K2 + Mm + k] = f2bf(w1);
}

// ---------- 2) h0 init ----------
__global__ __launch_bounds__(256) void init_h(const float* __restrict__ NATree,
                                              float* __restrict__ hF,
                                              __bf16* __restrict__ hB) {
    int idx = blockIdx.x * 256 + threadIdx.x;     // over Nn*Mm
    if (idx >= Nn * Mm) return;
    int n = idx / Mm, m = idx % Mm;
    float v = NATree[(size_t)n * (Ll * Mm) + (size_t)(Ll - 1) * Mm + m];
    hF[idx] = v;
    hB[idx] = f2bf(v);
}

// ---------- 3) one scan step: h_out = tanh([r|h] @ [W0^T;W1^T] + b) + r ----------
__global__ __launch_bounds__(256) void step_kernel(const float* __restrict__ rPtr,   // f32, row stride rStride
                                                   long long rStride,
                                                   const __bf16* __restrict__ hIn,   // bf16, row stride Mm
                                                   const __bf16* __restrict__ Wb,    // [Mm][K2] bf16
                                                   const float* __restrict__ bias,   // [Mm]
                                                   float* __restrict__ hOutF,
                                                   __bf16* __restrict__ hOutB) {
    __shared__ __bf16 As[2][BM][PITCH];
    __shared__ __bf16 Bs[2][BN][PITCH];

    const int tid  = threadIdx.x;
    const int row0 = blockIdx.y * BM;
    const int col0 = blockIdx.x * BN;

    const int wv   = tid >> 5;
    const int lane = tid & 31;
    const int wr   = wv & 3;          // 4 row groups of 32
    const int wc   = wv >> 2;         // 2 col groups of 64
    const int loE  = (lane < 16) ? 0 : 8;   // fragment element offset (ISA 16-bit A/B layout)
    const int nIdx = lane & 15;
    const int mAdd = (lane < 16) ? 0 : 8;

    v8f acc[2][4];
#pragma unroll
    for (int i = 0; i < 2; ++i)
#pragma unroll
        for (int j = 0; j < 4; ++j)
            acc[i][j] = (v8f){0.f,0.f,0.f,0.f,0.f,0.f,0.f,0.f};

    // ---- tile loaders (straight-line, branch decided once per chunk) ----
    auto loadB = [&](int kc, int buf) {
        const int k0 = kc * BK;
#pragma unroll
        for (int i = 0; i < 4; ++i) {
            int lin = tid + i * 256;              // 1024 items: 128 cols x 8 quads of 4 bf16
            int j = lin >> 3;
            int q = lin & 7;
            const __bf16* g = Wb + (size_t)(col0 + j) * K2 + k0 + q * 4;
            __bf16* l = &Bs[buf][j][q * 4];
#if ASYNC_LDS
            async_cp_b64(g, l);
#else
            *(uint2*)l = *(const uint2*)g;
#endif
        }
    };
    auto loadA = [&](int kc, int buf) {
        const int k0 = kc * BK;
        if (k0 < Mm) {
            // r-phase: f32 source, convert to bf16 on the fly (VMEM loads batched, then cvt+ds_store)
            float2 f[8];
#pragma unroll
            for (int i = 0; i < 8; ++i) {
                int lin = tid + i * 256;          // 2048 items: 128 rows x 16 pairs
                int row = lin >> 4;
                int pr  = lin & 15;
                f[i] = *(const float2*)(rPtr + (size_t)(row0 + row) * rStride + (k0 + pr * 2));
            }
#pragma unroll
            for (int i = 0; i < 8; ++i) {
                int lin = tid + i * 256;
                int row = lin >> 4;
                int pr  = lin & 15;
                *(uint32_t*)&As[buf][row][pr * 2] = pack2bf(f[i].x, f[i].y);
            }
            // prefetch one chunk further ahead of the f32 stream
            if (k0 + 2 * BK < Mm)
                __builtin_prefetch(rPtr + (size_t)(row0 + (tid >> 1)) * rStride + k0 + 2 * BK, 0, 1);
        } else {
            // h-phase: bf16 source, straight copy
#pragma unroll
            for (int i = 0; i < 4; ++i) {
                int lin = tid + i * 256;          // 1024 items: 128 rows x 8 quads of 4 bf16
                int row = lin >> 3;
                int q   = lin & 7;
                const __bf16* g = hIn + (size_t)(row0 + row) * Mm + (k0 - Mm) + q * 4;
                __bf16* l = &As[buf][row][q * 4];
#if ASYNC_LDS
                async_cp_b64(g, l);
#else
                *(uint2*)l = *(const uint2*)g;
#endif
            }
        }
    };

    const int NK = K2 / BK;           // 72 K-chunks; first 36 read r (f32), rest read h (bf16)
    loadB(0, 0);
    loadA(0, 0);

    for (int kc = 0; kc < NK; ++kc) {
        const int buf = kc & 1;
#if ASYNC_LDS
        wait_async0();                // async fills of buf must land before the barrier
#endif
        __syncthreads();

        if (kc + 1 < NK) {            // overlap next chunk's fill with this chunk's WMMA
            loadB(kc + 1, buf ^ 1);
            loadA(kc + 1, buf ^ 1);
        }

        // ---- WMMA: 2 row-tiles x 4 col-tiles of 16x16, K=32 ----
        v16bf aF[2], bF[4];
#pragma unroll
        for (int ti = 0; ti < 2; ++ti) {
            int r = wr * 32 + ti * 16 + (lane & 15);
            aF[ti] = ld_frag(&As[buf][r][loE]);
        }
#pragma unroll
        for (int tj = 0; tj < 4; ++tj) {
            int c = wc * 64 + tj * 16 + (lane & 15);
            bF[tj] = ld_frag(&Bs[buf][c][loE]);
        }
#pragma unroll
        for (int ti = 0; ti < 2; ++ti)
#pragma unroll
            for (int tj = 0; tj < 4; ++tj)
                acc[ti][tj] = __builtin_amdgcn_wmma_f32_16x16x32_bf16(
                    false, aF[ti], false, bF[tj], (short)0, acc[ti][tj], false, false);
    }

    // ---- epilogue: +bias, tanh, +r, store f32 + bf16 ----
#pragma unroll
    for (int ti = 0; ti < 2; ++ti) {
#pragma unroll
        for (int tj = 0; tj < 4; ++tj) {
            int gColBase = col0 + wc * 64 + tj * 16 + nIdx;
            float b = bias[gColBase];
#pragma unroll
            for (int v = 0; v < 8; ++v) {
                int gRow = row0 + wr * 32 + ti * 16 + v + mAdd;
                float cv = acc[ti][tj][v] + b;
                float hv = tanhf(cv) + rPtr[(size_t)gRow * rStride + gColBase];
                hOutF[(size_t)gRow * Mm + gColBase] = hv;
                hOutB[(size_t)gRow * Mm + gColBase] = f2bf(hv);
            }
        }
    }
}

// ---------- 4) res = tanh(tanh(h)+x); 129-tap valid conv ----------
__global__ __launch_bounds__(256) void final_kernel(const float* __restrict__ hF,
                                                    const float* __restrict__ x,
                                                    const float* __restrict__ c2w,
                                                    const float* __restrict__ c2b,
                                                    float* __restrict__ out) {
    __shared__ float res[Mm];
    __shared__ float w[KC];
    const int n = blockIdx.x;
    for (int i = threadIdx.x; i < Mm; i += 256)
        res[i] = tanhf(tanhf(hF[(size_t)n * Mm + i]) + x[(size_t)n * Mm + i]);
    for (int i = threadIdx.x; i < KC; i += 256)
        w[i] = c2w[i];
    __syncthreads();
    const float b = c2b[0];
    for (int o = threadIdx.x; o < OUTW; o += 256) {
        float s = 0.f;
        for (int k = 0; k < KC; ++k)
            s = fmaf(res[o + k], w[k], s);
        out[(size_t)n * OUTW + o] = s + b;
    }
}

// ---------- host orchestration ----------
extern "C" void kernel_launch(void* const* d_in, const int* in_sizes, int n_in,
                              void* d_out, int out_size, void* d_ws, size_t ws_size,
                              hipStream_t stream) {
    const float* NATree = (const float*)d_in[0];
    const float* x      = (const float*)d_in[1];
    const float* convW  = (const float*)d_in[2];
    const float* convb  = (const float*)d_in[3];
    const float* c2w    = (const float*)d_in[4];
    const float* c2b    = (const float*)d_in[5];
    float* out          = (float*)d_out;

    // workspace carve-up (256B aligned)
    char* ws = (char*)d_ws;
    size_t off = 0;
    auto carve = [&](size_t bytes) {
        void* p = ws + off;
        off = (off + bytes + 255) & ~(size_t)255;
        return p;
    };
    __bf16* Wb   = (__bf16*)carve((size_t)Mm * K2 * sizeof(__bf16));   // 5.3 MB
    float*  hF0  = (float*) carve((size_t)Nn * Mm * sizeof(float));    // 4.7 MB
    float*  hF1  = (float*) carve((size_t)Nn * Mm * sizeof(float));    // 4.7 MB
    __bf16* hB0  = (__bf16*)carve((size_t)Nn * Mm * sizeof(__bf16));   // 2.4 MB
    __bf16* hB1  = (__bf16*)carve((size_t)Nn * Mm * sizeof(__bf16));   // 2.4 MB
    float*  hF[2]  = { hF0, hF1 };
    __bf16* hB[2]  = { hB0, hB1 };

    prep_w<<<(Mm * Mm + 255) / 256, 256, 0, stream>>>(convW, Wb);
    init_h<<<(Nn * Mm + 255) / 256, 256, 0, stream>>>(NATree, hF0, hB0);

    dim3 grid(Mm / BN, Nn / BM);   // 9 x 8 blocks
    int cur = 0;
    for (int t = 0; t < 31; ++t) {
        int s = 30 - t;                                   // scan uses flipped slices 30..0
        const float* rPtr = NATree + (size_t)s * Mm;      // rows strided by L*M
        step_kernel<<<grid, 256, 0, stream>>>(rPtr, (long long)(Ll * Mm),
                                              hB[cur], Wb, convb,
                                              hF[1 - cur], hB[1 - cur]);
        cur ^= 1;
    }
    final_kernel<<<Nn, 256, 0, stream>>>(hF[cur], x, c2w, c2b, out);
}
